// TabNet_46093589021029
// MI455X (gfx1250) — matrix operand order repaired and checked by reading, not compiled, not run on hardware
//
#include <hip/hip_runtime.h>
#include <math.h>

typedef __bf16 bf16;
typedef __bf16 v16bf __attribute__((ext_vector_type(16)));
typedef __bf16 bfx8  __attribute__((ext_vector_type(8)));
typedef float  v8f   __attribute__((ext_vector_type(8)));

#define BB 32768
#define FF 256
#define HH 128
#define OUTD 64
#define SS 6
#define SQH 0.70710678118654752f

static __host__ int cdiv(int a, int b) { return (a + b - 1) / b; }

// ---------------- weight convert (f32 KxN row-major -> bf16 NxK "transposed") ----
__global__ void k_cvt_w(const float* __restrict__ W, bf16* __restrict__ Wt,
                        int K, int N, int count) {
    int total = K * N * count;
    for (int idx = blockIdx.x * blockDim.x + threadIdx.x; idx < total;
         idx += gridDim.x * blockDim.x) {
        int s = idx / (K * N);
        int r = idx - s * (K * N);
        int n = r / K, k = r - n * K;
        Wt[idx] = (bf16)W[(size_t)s * K * N + (size_t)k * N + n];
    }
}

// ---------------- init: complement mask = 1, entropy accum = 0 -------------------
__global__ void k_init(float* __restrict__ comp, float* __restrict__ ent) {
    int idx = blockIdx.x * blockDim.x + threadIdx.x;
    if (idx < BB * FF) comp[idx] = 1.0f;
    if (idx == 0) ent[0] = 0.0f;
}

// ---------------- per-column partial sums for raw x (256 cols, 128 chunks) -------
__global__ __launch_bounds__(256) void k_colstats(const float* __restrict__ Hp,
                                                  float* __restrict__ partial) {
    int col = threadIdx.x;           // 0..255
    int chunk = blockIdx.x;          // 0..127
    size_t r0 = (size_t)chunk * 256;
    float s = 0.f, s2 = 0.f;
    for (int r = 0; r < 256; r++) {
        float v = Hp[(r0 + r) * 256 + col];
        s += v; s2 += v * v;
    }
    partial[chunk * 512 + col] = s;
    partial[chunk * 512 + 256 + col] = s2;
}

// ---------------- reduce per-chunk partials -> mean/rstd -------------------------
__global__ void k_finalize(const float* __restrict__ partial,
                           float* __restrict__ mean, float* __restrict__ rstd,
                           int nchunks) {
    int col = threadIdx.x;           // 256 threads
    float s = 0.f, s2 = 0.f;
    for (int c = 0; c < nchunks; c++) {
        s += partial[c * 512 + col];
        s2 += partial[c * 512 + 256 + col];
    }
    float m = s * (1.0f / (float)BB);
    float v = s2 * (1.0f / (float)BB) - m * m;
    mean[col] = m;
    rstd[col] = rsqrtf(v + 1e-5f);
}

// ---------------- encoder BN apply: xbn f32 + bf16 copy as initial "masked" ------
__global__ void k_encbn(const float* __restrict__ x, const float* __restrict__ mean,
                        const float* __restrict__ rstd, const float* __restrict__ g,
                        const float* __restrict__ b, float* __restrict__ xbn,
                        bf16* __restrict__ mbf) {
    int idx = blockIdx.x * blockDim.x + threadIdx.x;
    if (idx >= BB * FF) return;
    int col = idx & 255;
    float v = (x[idx] - mean[col]) * rstd[col] * g[col] + b[col];
    xbn[idx] = v;
    mbf[idx] = (bf16)v;
}

// ---------------- bf16 WMMA GEMM + fused per-column sum/sumsq partials -----------
// C[B x 256] = A[B x K] * W[K x 256]. A bf16 row-major (stride K); Wt bf16
// N-major (stride K). 8 waves/block, block tile 64x256, wave tile 32x64:
// 2 row-tiles x 4 col-tiles of v_wmma_f32_16x16x32_bf16 (64 accum VGPRs).
// Epilogue emits block-level column partials (deterministic LDS staging).
__global__ __launch_bounds__(256) void k_gemm(const bf16* __restrict__ A,
                                              const bf16* __restrict__ Wt,
                                              float* __restrict__ C,
                                              float* __restrict__ partial, int K) {
    __shared__ float smem[8][128];   // per wave: 64 cols x {sum, sumsq}

    const int wave = threadIdx.x >> 5;
    const int lane = threadIdx.x & 31;
    const int mwave = wave & 1;      // 2 row groups of 32 rows
    const int nwave = wave >> 1;     // 4 col groups of 64 cols
    const int m0 = blockIdx.x * 64 + 32 * mwave;
    const int n0 = 64 * nwave;
    const int row = lane & 15;
    const int half = lane >> 4;      // 0 or 1
    const int li = lane & 15;

    v8f acc[2][4];
#pragma unroll
    for (int mi = 0; mi < 2; mi++)
#pragma unroll
        for (int t = 0; t < 4; t++)
#pragma unroll
            for (int r = 0; r < 8; r++) acc[mi][t][r] = 0.0f;

    // A layout: lanes 0-15 hold K {0..7,16..23}; lanes 16-31 hold K {8..15,24..31}
    const bf16* Abase0 = A + (size_t)(m0 + row) * K + half * 8;
    const bf16* Abase1 = A + (size_t)(m0 + 16 + row) * K + half * 8;

    for (int k0 = 0; k0 < K; k0 += 32) {
        v16bf af[2];
        {
            bfx8 lo0 = *(const bfx8*)(Abase0 + k0);
            bfx8 hi0 = *(const bfx8*)(Abase0 + k0 + 16);
            bfx8 lo1 = *(const bfx8*)(Abase1 + k0);
            bfx8 hi1 = *(const bfx8*)(Abase1 + k0 + 16);
#pragma unroll
            for (int i = 0; i < 8; i++) {
                af[0][i] = lo0[i]; af[0][i + 8] = hi0[i];
                af[1][i] = lo1[i]; af[1][i + 8] = hi1[i];
            }
        }
#pragma unroll
        for (int t = 0; t < 4; t++) {
            // B layout: lane n%16 holds col n; lanes 0-15 K {0..7,16..23}, lanes 16-31 K {8..15,24..31}
            const bf16* Bb = Wt + (size_t)(n0 + t * 16 + li) * K + k0 + half * 8;
            bfx8 blo = *(const bfx8*)(Bb);
            bfx8 bhi = *(const bfx8*)(Bb + 16);
            v16bf bfr;
#pragma unroll
            for (int i = 0; i < 8; i++) { bfr[i] = blo[i]; bfr[i + 8] = bhi[i]; }
            acc[0][t] = __builtin_amdgcn_wmma_f32_16x16x32_bf16(
                false, af[0], false, bfr, (short)0, acc[0][t], false, false);
            acc[1][t] = __builtin_amdgcn_wmma_f32_16x16x32_bf16(
                false, af[1], false, bfr, (short)0, acc[1][t], false, false);
        }
    }

    // Store C. C/D layout: vgpr r -> row (half*8 + r), col = lane&15.
#pragma unroll
    for (int mi = 0; mi < 2; mi++) {
        const int crow0 = m0 + 16 * mi + half * 8;
#pragma unroll
        for (int t = 0; t < 4; t++)
#pragma unroll
            for (int r = 0; r < 8; r++)
                C[(size_t)(crow0 + r) * 256 + n0 + t * 16 + li] = acc[mi][t][r];
    }

    // Fused column stats over this block's 64 rows (deterministic, no atomics).
#pragma unroll
    for (int t = 0; t < 4; t++) {
        float s = 0.f, s2 = 0.f;
#pragma unroll
        for (int mi = 0; mi < 2; mi++)
#pragma unroll
            for (int r = 0; r < 8; r++) {
                float v = acc[mi][t][r];
                s += v; s2 += v * v;
            }
        // combine the two 16-row halves (lanes L and L+16 hold same column)
        s  += __shfl_xor(s, 16, 32);
        s2 += __shfl_xor(s2, 16, 32);
        if (half == 0) {
            smem[wave][t * 32 + li * 2 + 0] = s;
            smem[wave][t * 32 + li * 2 + 1] = s2;
        }
    }
    __syncthreads();
    // thread t reduces column t across the 2 row-group waves of its col group
    {
        int c = threadIdx.x;                  // 0..255
        int ng = c >> 6;                      // col group 0..3
        int cl = c & 63;
        int sl = (cl >> 4) * 32 + (cl & 15) * 2;
        int w0 = ng * 2, w1 = ng * 2 + 1;
        float s  = smem[w0][sl + 0] + smem[w1][sl + 0];
        float s2 = smem[w0][sl + 1] + smem[w1][sl + 1];
        partial[blockIdx.x * 512 + c] = s;
        partial[blockIdx.x * 512 + 256 + c] = s2;
    }
}

// ---------------- BN + GLU (+optional residual*sqrt(1/2)), emit f32/bf16/fc ------
__global__ void k_bnglu(const float* __restrict__ hpre, const float* __restrict__ mean,
                        const float* __restrict__ rstd, const float* __restrict__ g,
                        const float* __restrict__ b, const float* __restrict__ resid,
                        float* __restrict__ hout, bf16* __restrict__ hbf,
                        bf16* __restrict__ fcbf) {
    int idx = blockIdx.x * blockDim.x + threadIdx.x;
    if (idx >= BB * HH) return;
    int col = idx & 127;
    size_t base = (size_t)(idx >> 7) * 256;
    float a  = (hpre[base + col] - mean[col]) * rstd[col] * g[col] + b[col];
    int c2 = col + 128;
    float sg = (hpre[base + c2] - mean[c2]) * rstd[c2] * g[c2] + b[c2];
    float v = a * (1.0f / (1.0f + expf(-sg)));
    if (resid) v = (v + resid[idx]) * SQH;
    hout[idx] = v;
    hbf[idx] = (bf16)v;
    if (fcbf && col >= OUTD) fcbf[(size_t)(idx >> 7) * OUTD + (col - OUTD)] = (bf16)v;
}

// ---------------- attention: BN(coef)*compl -> softmax -> mask/entropy/compl -----
__global__ __launch_bounds__(256) void k_mask(const float* __restrict__ coefpre,
                                              const float* __restrict__ mean,
                                              const float* __restrict__ rstd,
                                              const float* __restrict__ g,
                                              const float* __restrict__ b,
                                              const float* __restrict__ xbn,
                                              float* __restrict__ comp,
                                              bf16* __restrict__ maskedbf,
                                              float* __restrict__ ent) {
    int wave = threadIdx.x >> 5, lane = threadIdx.x & 31;
    int rowi = blockIdx.x * 8 + wave;             // one wave32 per row, 32 lanes x 8 cols
    size_t base = (size_t)rowi * 256;
    float vals[8], cms[8];
    float mx = -3.4e38f;
#pragma unroll
    for (int j = 0; j < 8; j++) {
        int c = lane + 32 * j;
        float bn = (coefpre[base + c] - mean[c]) * rstd[c] * g[c] + b[c];
        float cm = comp[base + c];
        float t = bn * cm;
        vals[j] = t; cms[j] = cm;
        mx = fmaxf(mx, t);
    }
#pragma unroll
    for (int o = 16; o >= 1; o >>= 1) mx = fmaxf(mx, __shfl_xor(mx, o, 32));
    float sum = 0.f;
#pragma unroll
    for (int j = 0; j < 8; j++) { vals[j] = expf(vals[j] - mx); sum += vals[j]; }
#pragma unroll
    for (int o = 16; o >= 1; o >>= 1) sum += __shfl_xor(sum, o, 32);
    float inv = 1.0f / sum, e = 0.f;
#pragma unroll
    for (int j = 0; j < 8; j++) {
        int c = lane + 32 * j;
        float m = vals[j] * inv;
        e += -m * logf(m + 1e-5f);
        comp[base + c] = cms[j] * (1.5f - m);
        maskedbf[base + c] = (bf16)(m * xbn[base + c]);
    }
#pragma unroll
    for (int o = 16; o >= 1; o >>= 1) e += __shfl_xor(e, o, 32);
    if (lane == 0) atomicAdd(ent, e * (1.0f / ((float)BB * (float)(SS - 1))));
}

// ---------------- output: 5*relu(xbn[:, :64]) and entropy scalar -----------------
__global__ void k_out(const float* __restrict__ xbn, const float* __restrict__ ent,
                      float* __restrict__ out) {
    int idx = blockIdx.x * blockDim.x + threadIdx.x;
    if (idx < BB * OUTD) {
        int col = idx & 63;
        size_t row = idx >> 6;
        float v = xbn[row * 256 + col];
        out[idx] = 5.0f * fmaxf(v, 0.0f);
    } else if (idx == BB * OUTD) {
        out[idx] = ent[0];
    }
}

extern "C" void kernel_launch(void* const* d_in, const int* in_sizes, int n_in,
                              void* d_out, int out_size, void* d_ws, size_t ws_size,
                              hipStream_t stream) {
    const float* x    = (const float*)d_in[0];
    const float* eg   = (const float*)d_in[1];
    const float* eb   = (const float*)d_in[2];
    const float* W1   = (const float*)d_in[3];
    const float* g1   = (const float*)d_in[4];
    const float* b1   = (const float*)d_in[5];
    const float* W2   = (const float*)d_in[6];
    const float* g2   = (const float*)d_in[7];
    const float* b2   = (const float*)d_in[8];
    const float* W3   = (const float*)d_in[9];
    const float* g3   = (const float*)d_in[10];
    const float* b3   = (const float*)d_in[11];
    const float* W4   = (const float*)d_in[12];
    const float* g4   = (const float*)d_in[13];
    const float* b4   = (const float*)d_in[14];
    const float* Wc   = (const float*)d_in[15];
    const float* gc   = (const float*)d_in[16];
    const float* bc   = (const float*)d_in[17];
    float* out = (float*)d_out;

    char* w = (char*)d_ws;
    auto alloc = [&](size_t bytes) -> char* {
        char* p = w;
        w += (bytes + 255) & ~(size_t)255;
        return p;
    };
    float* xbn      = (float*)alloc((size_t)BB * FF * 4);   // 33.5 MB
    float* hpre     = (float*)alloc((size_t)BB * 256 * 4);  // 33.5 MB (also coef-pre)
    float* h        = (float*)alloc((size_t)BB * HH * 4);   // 16.7 MB
    float* comp     = (float*)alloc((size_t)BB * FF * 4);   // 33.5 MB
    bf16*  maskedbf = (bf16*)alloc((size_t)BB * FF * 2);    // 16.8 MB
    bf16*  hbf      = (bf16*)alloc((size_t)BB * HH * 2);    //  8.4 MB
    bf16*  fcbf     = (bf16*)alloc((size_t)BB * OUTD * 2);  //  4.2 MB
    bf16*  W1t      = (bf16*)alloc((size_t)256 * 256 * 2);
    bf16*  W2t      = (bf16*)alloc((size_t)128 * 256 * 2);
    bf16*  W3t      = (bf16*)alloc((size_t)SS * 128 * 256 * 2);
    bf16*  W4t      = (bf16*)alloc((size_t)SS * 128 * 256 * 2);
    bf16*  Wct      = (bf16*)alloc((size_t)(SS - 1) * 64 * 256 * 2);
    float* partial  = (float*)alloc((size_t)512 * 512 * 4); // 1 MB (gemm partials)
    float* mean     = (float*)alloc(256 * 4);
    float* rstd     = (float*)alloc(256 * 4);
    float* entp     = (float*)alloc(256);

    // weights -> bf16 transposed (N-major)
    k_cvt_w<<<cdiv(256 * 256, 256), 256, 0, stream>>>(W1, W1t, 256, 256, 1);
    k_cvt_w<<<cdiv(128 * 256, 256), 256, 0, stream>>>(W2, W2t, 128, 256, 1);
    k_cvt_w<<<cdiv(SS * 128 * 256, 256), 256, 0, stream>>>(W3, W3t, 128, 256, SS);
    k_cvt_w<<<cdiv(SS * 128 * 256, 256), 256, 0, stream>>>(W4, W4t, 128, 256, SS);
    k_cvt_w<<<cdiv((SS - 1) * 64 * 256, 256), 256, 0, stream>>>(Wc, Wct, 64, 256, SS - 1);

    k_init<<<cdiv(BB * FF, 256), 256, 0, stream>>>(comp, entp);

    // encoder BN (stats on raw x need a standalone pass)
    k_colstats<<<128, 256, 0, stream>>>(x, partial);
    k_finalize<<<1, 256, 0, stream>>>(partial, mean, rstd, 128);
    k_encbn<<<cdiv(BB * FF, 256), 256, 0, stream>>>(x, mean, rstd, eg, eb, xbn, maskedbf);

    const int gemmGrid = BB / 64;         // 512 blocks
    const int bngluGrid = cdiv(BB * HH, 256);

    for (int ni = 0; ni < SS; ni++) {
        // stage 1: masked @ W1 (K=256)
        k_gemm<<<gemmGrid, 256, 0, stream>>>(maskedbf, W1t, hpre, partial, 256);
        k_finalize<<<1, 256, 0, stream>>>(partial, mean, rstd, 512);
        k_bnglu<<<bngluGrid, 256, 0, stream>>>(hpre, mean, rstd, g1, b1,
                                               nullptr, h, hbf, nullptr);
        // stage 2: h @ W2 (K=128) + residual
        k_gemm<<<gemmGrid, 256, 0, stream>>>(hbf, W2t, hpre, partial, 128);
        k_finalize<<<1, 256, 0, stream>>>(partial, mean, rstd, 512);
        k_bnglu<<<bngluGrid, 256, 0, stream>>>(hpre, mean, rstd, g2, b2,
                                               h, h, hbf, nullptr);
        // stage 3: h @ W3[ni] + residual
        k_gemm<<<gemmGrid, 256, 0, stream>>>(hbf, W3t + (size_t)ni * 128 * 256,
                                             hpre, partial, 128);
        k_finalize<<<1, 256, 0, stream>>>(partial, mean, rstd, 512);
        k_bnglu<<<bngluGrid, 256, 0, stream>>>(hpre, mean, rstd, g3 + ni * 256,
                                               b3 + ni * 256, h, h, hbf, nullptr);
        // stage 4: h @ W4[ni] + residual, extract fc = h[:, 64:]
        k_gemm<<<gemmGrid, 256, 0, stream>>>(hbf, W4t + (size_t)ni * 128 * 256,
                                             hpre, partial, 128);
        k_finalize<<<1, 256, 0, stream>>>(partial, mean, rstd, 512);
        k_bnglu<<<bngluGrid, 256, 0, stream>>>(hpre, mean, rstd, g4 + ni * 256,
                                               b4 + ni * 256, h, h, hbf, fcbf);
        // attention stage
        if (ni < SS - 1) {
            k_gemm<<<gemmGrid, 256, 0, stream>>>(fcbf, Wct + (size_t)ni * 64 * 256,
                                                 hpre, partial, 64);
            k_finalize<<<1, 256, 0, stream>>>(partial, mean, rstd, 512);
            k_mask<<<BB / 8, 256, 0, stream>>>(hpre, mean, rstd, gc + ni * 256,
                                               bc + ni * 256, xbn, comp, maskedbf, entp);
        }
    }

    k_out<<<cdiv(BB * OUTD + 1, 256), 256, 0, stream>>>(xbn, entp, out);
}
